// NTXentLoss_76802605187187
// MI455X (gfx1250) — matrix operand order repaired and checked by reading, not compiled, not run on hardware
//
#include <hip/hip_runtime.h>
#include <hip/hip_bf16.h>
#include <math.h>

typedef __attribute__((ext_vector_type(16))) _Float16 v16h;
typedef __attribute__((ext_vector_type(8)))  _Float16 v8h;
typedef __attribute__((ext_vector_type(4)))  _Float16 v4h;
typedef __attribute__((ext_vector_type(8)))  float    v8f;

#define DIMS   128
#define INV_T  14.285714285714286f   // 1/0.07
#define SMAX   14.285714285714286f   // upper bound of sim/T (cosine <= 1)
#define THRESH 0.1f
#define WAVES  8

// ---------------------------------------------------------------------------
// accumulator init (d_ws is poisoned; must be zeroed deterministically per call)
__global__ void ntx_zero_acc(float* acc) { acc[0] = 0.0f; }

// ---------------------------------------------------------------------------
// Row-normalize f32 features -> f16 matrix in workspace. One wave per row:
// 32 lanes x float4 = 128 elements; shfl-xor tree for ||x||^2.
__global__ __launch_bounds__(256) void ntx_normalize_f16(
    const float* __restrict__ feats, _Float16* __restrict__ fmat, int n) {
  const int lane = threadIdx.x & 31;
  const int wave = threadIdx.x >> 5;
  const int row  = blockIdx.x * WAVES + wave;
  if (row >= n) return;
  const float4 v = reinterpret_cast<const float4*>(feats + (size_t)row * DIMS)[lane];
  float ss = v.x * v.x + v.y * v.y + v.z * v.z + v.w * v.w;
#pragma unroll
  for (int off = 16; off >= 1; off >>= 1) ss += __shfl_xor(ss, off);
  const float scale = 1.0f / fmaxf(sqrtf(ss), 1e-8f);
  v4h h;
  h[0] = (_Float16)(v.x * scale);
  h[1] = (_Float16)(v.y * scale);
  h[2] = (_Float16)(v.z * scale);
  h[3] = (_Float16)(v.w * scale);
  *reinterpret_cast<v4h*>(fmat + (size_t)row * DIMS + lane * 4) = h;
}

// ---------------------------------------------------------------------------
// Fused sim-GEMM + branchless fixed-shift softmax-denominator + positive-pair
// accumulation. One workgroup per 16-row tile; 8 waves sweep the columns in
// 16-wide tiles. A fragments (16 rows x K128) stay in registers all sweep.
// Since cosine sim <= 1, exp(s - SMAX) is exactly representable in f32 with
// no overflow -> no online max, no divergent branches in the hot loop.
__global__ __launch_bounds__(256) void ntx_main(
    const _Float16* __restrict__ fmat, const float* __restrict__ labels,
    float* __restrict__ acc, int n) {
  __shared__ float red[WAVES][16][4];

  const int lane = threadIdx.x & 31;
  const int wave = threadIdx.x >> 5;
  const int lidx = lane & 15;   // column index inside tile / row-provider
  const int half = lane >> 4;   // 0: lanes 0-15, 1: lanes 16-31
  const int r0   = blockIdx.x * 16;

  // ---- A fragments: 16-bit A 16x32 layout (ISA 7.12.2):
  //   lanes 0-15 : halves[0..7]=K0..7,  halves[8..15]=K16..23
  //   lanes 16-31: halves[0..7]=K8..15, halves[8..15]=K24..31
  const _Float16* arow = fmat + (size_t)(r0 + lidx) * DIMS;
  const int aklo = half ? 8 : 0;
  v16h afrag[4];
#pragma unroll
  for (int kc = 0; kc < 4; ++kc) {
    const v8h lo = *reinterpret_cast<const v8h*>(arow + kc * 32 + aklo);
    const v8h hi = *reinterpret_cast<const v8h*>(arow + kc * 32 + 16 + aklo);
    v16h a;
#pragma unroll
    for (int i = 0; i < 8; ++i) { a[i] = lo[i]; a[i + 8] = hi[i]; }
    afrag[kc] = a;
  }

  // labels of the 8 rows this lane's C fragment touches (M = half*8 + v)
  float rowLab[8];
#pragma unroll
  for (int v = 0; v < 8; ++v) rowLab[v] = labels[r0 + half * 8 + v];

  float sm[8], ps[8], pc[8];
#pragma unroll
  for (int v = 0; v < 8; ++v) { sm[v] = 0.f; ps[v] = 0.f; pc[v] = 0.f; }

  // ---- B fragment: 16-bit B 32x16 layout: lane n holds column n,
  //   lanes 0-15 -> K0..7,K16..23; lanes 16-31 -> K8..15,K24..31 (matches A).
  const int bkofs = half ? 8 : 0;
  const int rbase = r0 + half * 8;   // this lane's first row
  for (int c0 = wave * 16; c0 < n; c0 += 16 * WAVES) {
    const _Float16* brow = fmat + (size_t)(c0 + lidx) * DIMS;
    v8f cacc = {};
#pragma unroll
    for (int kc = 0; kc < 4; ++kc) {
      const v8h lo = *reinterpret_cast<const v8h*>(brow + kc * 32 + bkofs);
      const v8h hi = *reinterpret_cast<const v8h*>(brow + kc * 32 + bkofs + 16);
      v16h b;
#pragma unroll
      for (int i = 0; i < 8; ++i) { b[i] = lo[i]; b[i + 8] = hi[i]; }
      cacc = __builtin_amdgcn_wmma_f32_16x16x32_f16(
          /*neg_a=*/false, afrag[kc], /*neg_b=*/false, b,
          /*c_mod=*/(short)0, cacc, /*reuse_a=*/false, /*reuse_b=*/false);
    }
    // C layout: VGPR v, this lane -> (M = half*8 + v, N = lidx)
    const int   col  = c0 + lidx;
    const float clab = labels[col];
    const int   dv   = col - rbase;  // diagonal hits element v == dv (if 0..7)
#pragma unroll
    for (int v = 0; v < 8; ++v) {
      const float s    = cacc[v] * INV_T;
      const float e    = __expf(s - SMAX);
      const bool  diag = (v == dv);
      sm[v] += diag ? 0.0f : e;                       // denominator sum
      const bool  pos  = (!diag) & (fabsf(clab - rowLab[v]) < THRESH);
      const float m    = pos ? 1.0f : 0.0f;
      pc[v] += m;
      ps[v] = fmaf(m, s, ps[v]);
    }
  }

  // additive merge across the 16 lanes of each half (disjoint column sets)
#pragma unroll
  for (int off = 1; off <= 8; off <<= 1) {
#pragma unroll
    for (int v = 0; v < 8; ++v) {
      sm[v] += __shfl_xor(sm[v], off);
      ps[v] += __shfl_xor(ps[v], off);
      pc[v] += __shfl_xor(pc[v], off);
    }
  }
  if (lidx == 0) {
#pragma unroll
    for (int v = 0; v < 8; ++v) {
      const int row = half * 8 + v;
      red[wave][row][0] = sm[v];
      red[wave][row][1] = ps[v];
      red[wave][row][2] = pc[v];
    }
  }
  __syncthreads();

  // cross-wave additive merge + per-row loss; threads 0-15 each own one row
  if (threadIdx.x < 16) {
    const int row = threadIdx.x;
    float S = 0.f, P = 0.f, C = 0.f;
#pragma unroll
    for (int w = 0; w < WAVES; ++w) {
      S += red[w][row][0];
      P += red[w][row][1];
      C += red[w][row][2];
    }
    const float log_denom = SMAX + __logf(S);
    float floss = P / C - log_denom;            // feature_losses[row]
#pragma unroll
    for (int off = 1; off <= 8; off <<= 1) floss += __shfl_xor(floss, off);
    if (row == 0) atomicAdd(acc, floss);
  }
}

// ---------------------------------------------------------------------------
__global__ void ntx_finalize(const float* __restrict__ acc, float* __restrict__ out, int n) {
  out[0] = -acc[0] / (float)n;
}

// ---------------------------------------------------------------------------
extern "C" void kernel_launch(void* const* d_in, const int* in_sizes, int n_in,
                              void* d_out, int out_size, void* d_ws, size_t ws_size,
                              hipStream_t stream) {
  const float* features = (const float*)d_in[0];
  const float* labels   = (const float*)d_in[1];
  const int n = in_sizes[1];  // 8192

  _Float16* fmat = (_Float16*)d_ws;                                  // n*128 f16 = 2 MB
  float*    acc  = (float*)((char*)d_ws + (size_t)n * DIMS * sizeof(_Float16));

  ntx_zero_acc<<<1, 1, 0, stream>>>(acc);
  ntx_normalize_f16<<<(n + WAVES - 1) / WAVES, 256, 0, stream>>>(features, fmat, n);
  ntx_main<<<n / 16, 256, 0, stream>>>(fmat, labels, acc, n);
  ntx_finalize<<<1, 1, 0, stream>>>(acc, (float*)d_out, n);
}
